// LSTM_model_3298534884111
// MI455X (gfx1250) — matrix operand order repaired and checked by reading, not compiled, not run on hardware
//
#include <hip/hip_runtime.h>
#include <hip/hip_bf16.h>
#include <math.h>

// ---------------------------------------------------------------------------
// 2-layer LSTM (B=128, T=256, I=512, H=1024) + FC(H->1) for MI455X / gfx1250.
// bf16 WMMA (v_wmma_f32_16x16x32_bf16) for all GEMMs; fused per-timestep
// kernel computes gates GEMM + bias + LSTM cell in registers (wave32 layout).
// ---------------------------------------------------------------------------

typedef __attribute__((ext_vector_type(16))) __bf16 v16bf;
typedef __attribute__((ext_vector_type(8)))  __bf16 v8bf;
typedef __attribute__((ext_vector_type(8)))  float  v8f;

#define DI static __device__ __forceinline__

constexpr int BATCH = 128;
constexpr int SEQ   = 256;
constexpr int IN0   = 512;
constexpr int HID   = 1024;
constexpr int GATES = 4 * HID;

// ---- WMMA fragment loaders (contiguous k order, same for A and B, wave32) ----------

// A-matrix 16x32 bf16 (MxK): lane m = lane%16; lanes<16 hold K = 0..15,
// lanes>=16 hold K = 16..31. Two contiguous 16B loads per lane.
DI v16bf load_A_frag(const __bf16* base, long row_stride, int lane) {
  const int m  = lane & 15;
  const int kb = (lane < 16) ? 0 : 16;
  const __bf16* p = base + (long)m * row_stride + kb;
  v8bf lo = *(const v8bf*)(p);        // K = kb .. kb+7
  v8bf hi = *(const v8bf*)(p + 8);    // K = kb+8 .. kb+15
  return __builtin_shufflevector(lo, hi, 0,1,2,3,4,5,6,7,8,9,10,11,12,13,14,15);
}

// B-matrix 32x16 bf16 (KxN), sourced from row-major W[N, K] (i.e. B = W^T):
// N = lane%16, element e -> K = e + (lane<16 ? 0 : 16). One 32B load per lane.
DI v16bf load_B_frag(const __bf16* base, long row_stride, int lane) {
  const int n    = lane & 15;
  const int koff = (lane < 16) ? 0 : 16;
  return *(const v16bf*)(base + (long)n * row_stride + koff);
}

DI v8f wmma_bf16(v16bf a, v16bf b, v8f c) {
  return __builtin_amdgcn_wmma_f32_16x16x32_bf16(
      /*neg_a=*/false, a, /*neg_b=*/false, b,
      /*c_mod=*/(short)0, c, /*reuse_a=*/false, /*reuse_b=*/false);
}

DI float sigmoid_f(float x) { return 1.0f / (1.0f + __expf(-x)); }

// ---------------------------------------------------------------------------
// Fused LSTM timestep.
//   xseq : [T, B, F]  bf16   (layer input sequence, time-major)
//   Wih  : [4H, F]    bf16 row-major
//   Whh  : [4H, H]    bf16 row-major
//   bias : [4H]       f32   (b_ih + b_hh combined)
//   cbuf : [B, H]     f32   (cell state, updated in place)
//   hseq : [T, B, H]  bf16  (this layer's output; h_{t-1} read from it)
// Grid: 64 blocks x 256 threads. Block -> 16 hidden columns; wave -> 16 rows.
// Each wave accumulates 4 tiles (i,f,g,o) for the SAME hidden columns, so the
// LSTM cell is evaluated entirely in registers after the K loop.
// ---------------------------------------------------------------------------
template <int F>
__global__ __launch_bounds__(256)
void lstm_step_kernel(const __bf16* __restrict__ xseq,
                      const __bf16* __restrict__ Wih,
                      const __bf16* __restrict__ Whh,
                      const float*  __restrict__ bias,
                      float*        __restrict__ cbuf,
                      __bf16*       __restrict__ hseq,
                      int t) {
  const int lane = threadIdx.x & 31;
  const int wave = threadIdx.x >> 5;      // 0..7
  const int m0   = wave * 16;             // row strip (batch)
  const int n0   = blockIdx.x * 16;       // hidden-column tile

  v8f acc_i = {}, acc_f = {}, acc_g = {}, acc_o = {};

  // ---- input projection: gates += x_t @ Wih^T  (K over F) ----
  {
    const __bf16* abase = xseq + ((long)t * BATCH + m0) * F;
    const long    rsA   = F;
#pragma unroll 2
    for (int k0 = 0; k0 < F; k0 += 32) {
      v16bf a  = load_A_frag(abase + k0, rsA, lane);
      v16bf bi = load_B_frag(Wih + (long)(n0 + 0 * HID) * F + k0, F, lane);
      v16bf bf = load_B_frag(Wih + (long)(n0 + 1 * HID) * F + k0, F, lane);
      v16bf bg = load_B_frag(Wih + (long)(n0 + 2 * HID) * F + k0, F, lane);
      v16bf bo = load_B_frag(Wih + (long)(n0 + 3 * HID) * F + k0, F, lane);
      acc_i = wmma_bf16(a, bi, acc_i);
      acc_f = wmma_bf16(a, bf, acc_f);
      acc_g = wmma_bf16(a, bg, acc_g);
      acc_o = wmma_bf16(a, bo, acc_o);
    }
  }

  // ---- recurrent projection: gates += h_{t-1} @ Whh^T (K over H) ----
  if (t > 0) {
    const __bf16* abase = hseq + ((long)(t - 1) * BATCH + m0) * HID;
#pragma unroll 2
    for (int k0 = 0; k0 < HID; k0 += 32) {
      v16bf a  = load_A_frag(abase + k0, HID, lane);
      v16bf bi = load_B_frag(Whh + (long)(n0 + 0 * HID) * HID + k0, HID, lane);
      v16bf bf = load_B_frag(Whh + (long)(n0 + 1 * HID) * HID + k0, HID, lane);
      v16bf bg = load_B_frag(Whh + (long)(n0 + 2 * HID) * HID + k0, HID, lane);
      v16bf bo = load_B_frag(Whh + (long)(n0 + 3 * HID) * HID + k0, HID, lane);
      acc_i = wmma_bf16(a, bi, acc_i);
      acc_f = wmma_bf16(a, bf, acc_f);
      acc_g = wmma_bf16(a, bg, acc_g);
      acc_o = wmma_bf16(a, bo, acc_o);
    }
  }

  // ---- LSTM cell, in registers (C/D layout: VGPR r -> M = r + (lane<16?0:8),
  //      N = lane%16; lanes>=16 carry rows 8..15 of the tile) ----
  const int   n  = n0 + (lane & 15);
  const float bi = bias[n + 0 * HID];
  const float bf = bias[n + 1 * HID];
  const float bg = bias[n + 2 * HID];
  const float bo = bias[n + 3 * HID];
  const int   mb = m0 + ((lane < 16) ? 0 : 8);

#pragma unroll
  for (int r = 0; r < 8; ++r) {
    const int m = mb + r;
    float iv = sigmoid_f(acc_i[r] + bi);
    float fv = sigmoid_f(acc_f[r] + bf);
    float gv = tanhf(acc_g[r] + bg);
    float ov = sigmoid_f(acc_o[r] + bo);
    float cp = (t == 0) ? 0.0f : cbuf[(long)m * HID + n];
    float cn = fv * cp + iv * gv;
    cbuf[(long)m * HID + n] = cn;
    hseq[((long)t * BATCH + m) * HID + n] = (__bf16)(ov * tanhf(cn));
  }
}

// ---- elementwise helpers ---------------------------------------------------

__global__ void f32_to_bf16_kernel(const float* __restrict__ src,
                                   __bf16* __restrict__ dst, long n) {
  long i = (long)blockIdx.x * blockDim.x + threadIdx.x;
  long stride = (long)gridDim.x * blockDim.x;
  for (; i < n; i += stride) dst[i] = (__bf16)src[i];
}

// x: [B,T,F] f32 -> xbf: [T,B,F] bf16 (time-major for the step kernel)
__global__ void transpose_x_kernel(const float* __restrict__ x,
                                   __bf16* __restrict__ xbf) {
  const long total = (long)BATCH * SEQ * IN0;
  long i = (long)blockIdx.x * blockDim.x + threadIdx.x;
  long stride = (long)gridDim.x * blockDim.x;
  for (; i < total; i += stride) {
    int f = (int)(i % IN0);
    long r = i / IN0;
    int b = (int)(r % BATCH);
    int t = (int)(r / BATCH);
    xbf[i] = (__bf16)x[((long)b * SEQ + t) * IN0 + f];
  }
}

__global__ void bias_combine_kernel(const float* __restrict__ a,
                                    const float* __restrict__ b,
                                    float* __restrict__ dst, int n) {
  int i = blockIdx.x * blockDim.x + threadIdx.x;
  if (i < n) dst[i] = a[i] + b[i];
}

// out[b] = h2[T-1, b, :] . Wfc + bfc   (hseq1 is [T,B,H] bf16)
__global__ void fc_kernel(const __bf16* __restrict__ hseq1,
                          const float* __restrict__ Wfc,
                          const float* __restrict__ bfc,
                          float* __restrict__ out) {
  int b = blockIdx.x * blockDim.x + threadIdx.x;
  if (b >= BATCH) return;
  const __bf16* h = hseq1 + ((long)(SEQ - 1) * BATCH + b) * HID;
  float s = 0.0f;
#pragma unroll 4
  for (int j = 0; j < HID; ++j) s += (float)h[j] * Wfc[j];
  out[b] = s + bfc[0];
}

// ---------------------------------------------------------------------------

extern "C" void kernel_launch(void* const* d_in, const int* in_sizes, int n_in,
                              void* d_out, int out_size, void* d_ws, size_t ws_size,
                              hipStream_t stream) {
  const float* x     = (const float*)d_in[0];
  const float* W_ih0 = (const float*)d_in[1];
  const float* W_hh0 = (const float*)d_in[2];
  const float* b_ih0 = (const float*)d_in[3];
  const float* b_hh0 = (const float*)d_in[4];
  const float* W_ih1 = (const float*)d_in[5];
  const float* W_hh1 = (const float*)d_in[6];
  const float* b_ih1 = (const float*)d_in[7];
  const float* b_hh1 = (const float*)d_in[8];
  const float* W_fc  = (const float*)d_in[9];
  const float* b_fc  = (const float*)d_in[10];
  float* out = (float*)d_out;

  // Workspace carve-up (256B aligned).
  size_t off = 0;
  auto carve = [&](size_t bytes) -> void* {
    void* p = (char*)d_ws + off;
    off += (bytes + 255) & ~(size_t)255;
    return p;
  };
  __bf16* xbf   = (__bf16*)carve(sizeof(__bf16) * (size_t)SEQ * BATCH * IN0);
  __bf16* wih0  = (__bf16*)carve(sizeof(__bf16) * (size_t)GATES * IN0);
  __bf16* whh0  = (__bf16*)carve(sizeof(__bf16) * (size_t)GATES * HID);
  __bf16* wih1  = (__bf16*)carve(sizeof(__bf16) * (size_t)GATES * HID);
  __bf16* whh1  = (__bf16*)carve(sizeof(__bf16) * (size_t)GATES * HID);
  float*  bias0 = (float*)carve(sizeof(float) * GATES);
  float*  bias1 = (float*)carve(sizeof(float) * GATES);
  __bf16* hseq0 = (__bf16*)carve(sizeof(__bf16) * (size_t)SEQ * BATCH * HID);
  __bf16* hseq1 = (__bf16*)carve(sizeof(__bf16) * (size_t)SEQ * BATCH * HID);
  float*  cbuf  = (float*)carve(sizeof(float) * (size_t)BATCH * HID);

  // ---- one-time (per call) conversions to bf16 ----
  transpose_x_kernel<<<2048, 256, 0, stream>>>(x, xbf);
  f32_to_bf16_kernel<<<2048, 256, 0, stream>>>(W_ih0, wih0, (long)GATES * IN0);
  f32_to_bf16_kernel<<<2048, 256, 0, stream>>>(W_hh0, whh0, (long)GATES * HID);
  f32_to_bf16_kernel<<<2048, 256, 0, stream>>>(W_ih1, wih1, (long)GATES * HID);
  f32_to_bf16_kernel<<<2048, 256, 0, stream>>>(W_hh1, whh1, (long)GATES * HID);
  bias_combine_kernel<<<(GATES + 255) / 256, 256, 0, stream>>>(b_ih0, b_hh0, bias0, GATES);
  bias_combine_kernel<<<(GATES + 255) / 256, 256, 0, stream>>>(b_ih1, b_hh1, bias1, GATES);

  const dim3 grid(HID / 16);   // 64 workgroups: 16 hidden cols each
  const dim3 block(256);       // 8 waves: 16 batch rows each

  // ---- layer 0 (F = 512) ----
  for (int t = 0; t < SEQ; ++t)
    lstm_step_kernel<IN0><<<grid, block, 0, stream>>>(xbf, wih0, whh0, bias0,
                                                      cbuf, hseq0, t);
  // ---- layer 1 (F = 1024) ----
  for (int t = 0; t < SEQ; ++t)
    lstm_step_kernel<HID><<<grid, block, 0, stream>>>(hseq0, wih1, whh1, bias1,
                                                      cbuf, hseq1, t);
  // ---- final FC ----
  fc_kernel<<<1, 128, 0, stream>>>(hseq1, W_fc, b_fc, out);
}
